// PhiMHA_77326591197284
// MI455X (gfx1250) — compile-verified, matrix operand order checked by reading, not run on hardware
//
#include <hip/hip_runtime.h>

// ---------------------------------------------------------------------------
// Phi MHA for MI455X (gfx1250, wave32, WMMA 16x16x32 f16 -> f32 accum)
//  B=1, S=2048, E=2560, H=32, D=80, rotary=32.  ~150 GFLOP -> matrix-core
//  bound.  All GEMM-shaped work on v_wmma_f32_16x16x32_f16; GEMM operands are
//  pre-converted to f16 once (halves HBM traffic), and LDS staging uses the
//  CDNA5 async path (global_load_async_to_lds_b128 + s_wait_asynccnt).
// ---------------------------------------------------------------------------

typedef __attribute__((ext_vector_type(16))) _Float16 v16h;
typedef __attribute__((ext_vector_type(8)))  _Float16 v8h;
typedef __attribute__((ext_vector_type(4)))  _Float16 v4h;
typedef __attribute__((ext_vector_type(8)))  float    v8f;
typedef __attribute__((ext_vector_type(4)))  float    v4f;

#define S_LEN 2048
#define EMB   2560
#define NH    32
#define HD    80
#define HDP   96   // head dim padded to multiple of 32 for K-dim of QK^T
#define RDIM  32

// Lane fragment slice per ISA 7.12.2: half h = lane>>4,
// elements 0..7 <- k = 8h..8h+7 ; elements 8..15 <- k = 16+8h..16+8h+7.
// Two contiguous 16-byte chunks -> two b128-class loads.
static __device__ __forceinline__ v16h frag_ld(const _Float16* row, int half) {
  v8h lo = *(const v8h*)(row + (half << 3));
  v8h hi = *(const v8h*)(row + 16 + (half << 3));
  v16h r;
#pragma unroll
  for (int i = 0; i < 8; ++i) { r[i] = lo[i]; r[i + 8] = hi[i]; }
  return r;
}

// CDNA5 async global->LDS copy, 16B per lane, tracked by ASYNCcnt.
// lds_off: 32-bit LDS byte offset (addrspace(3) value); gvoff: unsigned byte
// offset from saddr (GVS mode).
static __device__ __forceinline__ void async_ld_b128(unsigned lds_off,
                                                     unsigned gvoff,
                                                     const void* base) {
  asm volatile("global_load_async_to_lds_b128 %0, %1, %2"
               :: "v"(lds_off), "v"(gvoff), "s"(base)
               : "memory");
}
static __device__ __forceinline__ void wait_asynccnt0() {
  asm volatile("s_wait_asynccnt 0" ::: "memory");
}

// ---------------------------------------------------------------------------
// One-shot fp32 -> f16 conversion (vectorized x4).
// ---------------------------------------------------------------------------
__global__ __launch_bounds__(256)
void cvt_f32_f16(const float* __restrict__ in, _Float16* __restrict__ out, int n4) {
  int i = blockIdx.x * 256 + threadIdx.x;
  if (i < n4) {
    v4f v = *(const v4f*)(in + (size_t)i * 4);
    v4h h;
#pragma unroll
    for (int j = 0; j < 4; ++j) h[j] = (_Float16)v[j];
    *(v4h*)(out + (size_t)i * 4) = h;
  }
}

// ---------------------------------------------------------------------------
// GEMM: C[M,N] = A[M,K] * B[N,K]^T + bias[N]   (f16 in, f32 out)
// 256 threads = 8 waves, 128x128 C tile; wave (wm,wn) owns 32x64
// (2 A-frags x 4 B-frags, 8 accumulators).  K stepped by 32; A/B tiles
// (128x32 f16 = 8KB each) staged via async global->LDS b128 copies.
// ---------------------------------------------------------------------------
__global__ __launch_bounds__(256)
void wmma_gemm_f16nt(const _Float16* __restrict__ A, const _Float16* __restrict__ B,
                     const float* __restrict__ bias, float* __restrict__ C,
                     int M, int N, int K) {
  __shared__ __align__(16) _Float16 As[128][32];
  __shared__ __align__(16) _Float16 Bs[128][32];

  const int tid  = threadIdx.x;
  const int wave = tid >> 5, lane = tid & 31;
  const int half = lane >> 4, l16 = lane & 15;
  const int wm = wave >> 1, wn = wave & 1;       // 4 x 2 wave grid
  const int m0 = blockIdx.y * 128, n0 = blockIdx.x * 128;

  const unsigned ldsA = (unsigned)(unsigned long long)(&As[0][0]);
  const unsigned ldsB = (unsigned)(unsigned long long)(&Bs[0][0]);

  v8f acc[8] = {};

  for (int k0 = 0; k0 < K; k0 += 32) {
    // ---- async stage 128x32 f16 of A and B (16B chunks, 2 per matrix) ----
#pragma unroll
    for (int i = 0; i < 2; ++i) {
      int c   = tid + i * 256;          // 0..511 chunk id (128 rows x 4)
      int row = c >> 2;
      int co  = (c & 3) << 4;           // byte offset within 64B row
      async_ld_b128(ldsA + row * 64 + co,
                    (unsigned)(((m0 + row) * K + k0) * 2) + co, A);
      async_ld_b128(ldsB + row * 64 + co,
                    (unsigned)(((n0 + row) * K + k0) * 2) + co, B);
    }
    if (k0 + 32 < K) {  // pull next K tile toward L2 (global_prefetch_b8)
      __builtin_prefetch(A + (size_t)(m0 + (tid >> 1)) * K + k0 + 32, 0, 1);
      __builtin_prefetch(B + (size_t)(n0 + (tid >> 1)) * K + k0 + 32, 0, 1);
    }
    wait_asynccnt0();
    __syncthreads();

    // ---- 8 WMMAs per wave per K step ----
    v16h a0 = frag_ld(&As[wm * 32 + l16][0], half);
    v16h a1 = frag_ld(&As[wm * 32 + 16 + l16][0], half);
#pragma unroll
    for (int c = 0; c < 4; ++c) {
      v16h bf = frag_ld(&Bs[wn * 64 + c * 16 + l16][0], half);
      acc[c] = __builtin_amdgcn_wmma_f32_16x16x32_f16(
          false, a0, false, bf, (short)0, acc[c], false, false);
      acc[4 + c] = __builtin_amdgcn_wmma_f32_16x16x32_f16(
          false, a1, false, bf, (short)0, acc[4 + c], false, false);
    }
    __syncthreads();
  }

  // C/D layout: element r <-> row r + 8*half, col = l16 (per 16-col chunk)
#pragma unroll
  for (int i = 0; i < 2; ++i) {
#pragma unroll
    for (int c = 0; c < 4; ++c) {
      int col = n0 + wn * 64 + c * 16 + l16;
      float bv = bias[col];
#pragma unroll
      for (int r = 0; r < 8; ++r) {
        int row = m0 + wm * 32 + i * 16 + r + 8 * half;
        C[(size_t)row * N + col] = acc[i * 4 + c][r] + bv;
      }
    }
  }
}

// ---------------------------------------------------------------------------
// RoPE + pack: qkv fp32 [s][3*2560] ->
//   Qf, Kf : f16 [h][s][96]  (rotary on first 32 dims, d=80..95 zero)
//   Vt     : f16 [h][d][s]   (transposed so PV B-fragments stream along s)
// ---------------------------------------------------------------------------
__global__ __launch_bounds__(96)
void rope_pack(const float* __restrict__ qkv, const float* __restrict__ cosT,
               const float* __restrict__ sinT, _Float16* __restrict__ Qf,
               _Float16* __restrict__ Kf, _Float16* __restrict__ Vt) {
  const int s = blockIdx.x, h = blockIdx.y, d = threadIdx.x;
  const float* base = qkv + (size_t)s * (3 * EMB) + h * HD;

  float qv = 0.f, kv = 0.f;
  if (d < HD) {
    qv = base[d];
    kv = base[EMB + d];
    if (d < RDIM) {
      int   j  = d & 15;
      float c  = cosT[s * 16 + j];
      float sn = sinT[s * 16 + j];
      float q1 = base[j],        q2 = base[j + 16];
      float k1 = base[EMB + j],  k2 = base[EMB + j + 16];
      qv = (d < 16) ? (q1 * c - q2 * sn) : (q1 * sn + q2 * c);
      kv = (d < 16) ? (k1 * c - k2 * sn) : (k1 * sn + k2 * c);
    }
  }
  size_t o = ((size_t)h * S_LEN + s) * HDP + d;
  Qf[o] = (_Float16)qv;
  Kf[o] = (_Float16)kv;
  if (d < HD) Vt[((size_t)h * HD + d) * S_LEN + s] = (_Float16)base[2 * EMB + d];
}

// ---------------------------------------------------------------------------
// Causal flash attention; one wave per 16-row query tile, f16 output so the
// out-projection consumes it directly.
// ---------------------------------------------------------------------------
__global__ __launch_bounds__(128)
void flash_attn(const _Float16* __restrict__ Qf, const _Float16* __restrict__ Kf,
                const _Float16* __restrict__ Vt, _Float16* __restrict__ O) {
  __shared__ __align__(16) _Float16 Ps[4][16][32];

  const int tid = threadIdx.x, wave = tid >> 5, lane = tid & 31;
  const int half = lane >> 4, l16 = lane & 15;
  const int h  = blockIdx.y;
  const int q0 = blockIdx.x * 64 + wave * 16;
  const float scale = 0.11180339887498949f;  // 1/sqrt(80)

  v16h qfr[3];
  const _Float16* qrow = Qf + ((size_t)h * S_LEN + q0 + l16) * HDP;
#pragma unroll
  for (int c = 0; c < 3; ++c) qfr[c] = frag_ld(qrow + c * 32, half);

  v8f   o[5] = {};
  float m[8], l[8];
#pragma unroll
  for (int r = 0; r < 8; ++r) { m[r] = -1e30f; l[r] = 0.f; }

  const int nj = (q0 + 15) / 32 + 1;
  for (int j = 0; j < nj; ++j) {
    v8f sc0 = {}, sc1 = {};
    const _Float16* krow0 = Kf + ((size_t)h * S_LEN + j * 32 + l16) * HDP;
    const _Float16* krow1 = krow0 + 16 * HDP;
#pragma unroll
    for (int c = 0; c < 3; ++c) {
      sc0 = __builtin_amdgcn_wmma_f32_16x16x32_f16(
          false, qfr[c], false, frag_ld(krow0 + c * 32, half), (short)0, sc0, false, false);
      sc1 = __builtin_amdgcn_wmma_f32_16x16x32_f16(
          false, qfr[c], false, frag_ld(krow1 + c * 32, half), (short)0, sc1, false, false);
    }

#pragma unroll
    for (int r = 0; r < 8; ++r) {
      int   qi  = q0 + r + 8 * half;
      int   s0i = j * 32 + l16, s1i = s0i + 16;
      float a0 = sc0[r] * scale + ((s0i > qi) ? -10000.f : 0.f);
      float a1 = sc1[r] * scale + ((s1i > qi) ? -10000.f : 0.f);

      float v = fmaxf(a0, a1);
      v = fmaxf(v, __shfl_xor(v, 1)); v = fmaxf(v, __shfl_xor(v, 2));
      v = fmaxf(v, __shfl_xor(v, 4)); v = fmaxf(v, __shfl_xor(v, 8));
      float mn    = fmaxf(m[r], v);
      float alpha = __expf(m[r] - mn);
      float p0 = __expf(a0 - mn), p1 = __expf(a1 - mn);
      float ps = p0 + p1;
      ps += __shfl_xor(ps, 1); ps += __shfl_xor(ps, 2);
      ps += __shfl_xor(ps, 4); ps += __shfl_xor(ps, 8);
      l[r] = l[r] * alpha + ps;
      m[r] = mn;
#pragma unroll
      for (int c = 0; c < 5; ++c) o[c][r] *= alpha;

      Ps[wave][r + 8 * half][l16]      = (_Float16)p0;
      Ps[wave][r + 8 * half][l16 + 16] = (_Float16)p1;
    }
    __threadfence_block();  // wave-local LDS RAW ordering (dscnt wait)

    v16h pa = frag_ld(&Ps[wave][l16][0], half);
    const _Float16* vrow = Vt + ((size_t)h * HD + l16) * S_LEN + j * 32;
#pragma unroll
    for (int c = 0; c < 5; ++c) {
      o[c] = __builtin_amdgcn_wmma_f32_16x16x32_f16(
          false, pa, false, frag_ld(vrow + (size_t)c * 16 * S_LEN, half),
          (short)0, o[c], false, false);
    }
  }

#pragma unroll
  for (int c = 0; c < 5; ++c) {
    int col = h * HD + c * 16 + l16;
#pragma unroll
    for (int r = 0; r < 8; ++r) {
      int row = q0 + r + 8 * half;
      O[(size_t)row * EMB + col] = (_Float16)(o[c][r] / l[r]);
    }
  }
}

// ---------------------------------------------------------------------------
extern "C" void kernel_launch(void* const* d_in, const int* in_sizes, int n_in,
                              void* d_out, int out_size, void* d_ws, size_t ws_size,
                              hipStream_t stream) {
  const float* hidden = (const float*)d_in[0];
  const float* Wqkv_w = (const float*)d_in[1];
  const float* Wqkv_b = (const float*)d_in[2];
  const float* out_w  = (const float*)d_in[3];
  const float* out_b  = (const float*)d_in[4];
  const float* cosT   = (const float*)d_in[5];
  const float* sinT   = (const float*)d_in[6];

  char* ws = (char*)d_ws;
  size_t off = 0;
  float*    qkv   = (float*)(ws + off);    off += (size_t)S_LEN * 3 * EMB * sizeof(float);
  _Float16* Qf    = (_Float16*)(ws + off); off += (size_t)NH * S_LEN * HDP * sizeof(_Float16);
  _Float16* Kf    = (_Float16*)(ws + off); off += (size_t)NH * S_LEN * HDP * sizeof(_Float16);
  _Float16* Vt    = (_Float16*)(ws + off); off += (size_t)NH * HD * S_LEN * sizeof(_Float16);
  _Float16* att16 = (_Float16*)(ws + off); off += (size_t)S_LEN * EMB * sizeof(_Float16);
  _Float16* hid16 = (_Float16*)(ws + off); off += (size_t)S_LEN * EMB * sizeof(_Float16);
  _Float16* wqkv16= (_Float16*)(ws + off); off += (size_t)3 * EMB * EMB * sizeof(_Float16);
  _Float16* outw16= (_Float16*)(ws + off); off += (size_t)EMB * EMB * sizeof(_Float16);

  // 0) one-shot f16 conversions of GEMM operands
  {
    int n4h = (S_LEN * EMB) / 4, n4w = (3 * EMB * EMB) / 4, n4o = (EMB * EMB) / 4;
    cvt_f32_f16<<<(n4h + 255) / 256, 256, 0, stream>>>(hidden, hid16, n4h);
    cvt_f32_f16<<<(n4w + 255) / 256, 256, 0, stream>>>(Wqkv_w, wqkv16, n4w);
    cvt_f32_f16<<<(n4o + 255) / 256, 256, 0, stream>>>(out_w, outw16, n4o);
  }

  // 1) QKV projection: (2048 x 2560) * (7680 x 2560)^T + b
  wmma_gemm_f16nt<<<dim3((3 * EMB) / 128, S_LEN / 128), 256, 0, stream>>>(
      hid16, wqkv16, Wqkv_b, qkv, S_LEN, 3 * EMB, EMB);

  // 2) rotary + f16 packing (Q,K padded to 96; V transposed)
  rope_pack<<<dim3(S_LEN, NH), 96, 0, stream>>>(qkv, cosT, sinT, Qf, Kf, Vt);

  // 3) causal flash attention (writes f16)
  flash_attn<<<dim3(S_LEN / 64, NH), 128, 0, stream>>>(Qf, Kf, Vt, att16);

  // 4) output projection: (2048 x 2560) * (2560 x 2560)^T + b
  wmma_gemm_f16nt<<<dim3(EMB / 128, S_LEN / 128), 256, 0, stream>>>(
      att16, outw16, out_b, (float*)d_out, S_LEN, EMB, EMB);
}
